// BinaryLSTMClassifier_74062416053110
// MI455X (gfx1250) — compile-verified, matrix operand order checked by reading, not run on hardware
//
#include <hip/hip_runtime.h>
#include <hip/hip_bf16.h>
#include <stdint.h>
#include <stddef.h>

// Problem sizes (fixed by the reference)
#define HH   128      // hidden
#define DD   2048     // input size
#define BB   64       // batch
#define TT   256      // seq len
#define G4   512      // 4*H  (gate dim)

typedef __attribute__((ext_vector_type(16))) __bf16        v16bf;
typedef __attribute__((ext_vector_type(8)))  float         v8f;
typedef __attribute__((ext_vector_type(4)))  unsigned int  u32x4;
typedef __attribute__((ext_vector_type(8)))  int           i32x8;
typedef __attribute__((ext_vector_type(4)))  int           i32x4;

struct FragBits { u32x4 lo, hi; };

static __device__ __forceinline__ unsigned short f2bf(float f) {
  unsigned u = __builtin_bit_cast(unsigned, f);
  u += 0x7FFFu + ((u >> 16) & 1u);          // round-to-nearest-even
  return (unsigned short)(u >> 16);
}
static __device__ __forceinline__ float bf2f(unsigned short h) {
  unsigned u = ((unsigned)h) << 16;
  return __builtin_bit_cast(float, u);
}
static __device__ __forceinline__ unsigned pk2(float a, float b) {
  return (unsigned)f2bf(a) | ((unsigned)f2bf(b) << 16);
}
// Load a 16-half bf16 WMMA fragment: 8 halfs at p, 8 halfs at p+16
// (matches CDNA5 16-bit A/B VGPR layout: per-lane K chunks of 8).
static __device__ __forceinline__ v16bf frag_ld(const unsigned short* p) {
  FragBits q;
  q.lo = *reinterpret_cast<const u32x4*>(p);
  q.hi = *reinterpret_cast<const u32x4*>(p + 16);
  return __builtin_bit_cast(v16bf, q);
}
static __device__ __forceinline__ float sigm(float x) { return 1.0f / (1.0f + __expf(-x)); }
static __device__ __forceinline__ float tanh_f(float x) { return 1.0f - 2.0f / (__expf(2.0f * x) + 1.0f); }
static __device__ __forceinline__ v8f wmma_bf16(v16bf a, v16bf b, v8f c) {
  return __builtin_amdgcn_wmma_f32_16x16x32_bf16(false, a, false, b, (short)0, c, false, false);
}

// ---------------------------------------------------------------------------
// Tensor Data Mover: 2D tile load, descriptor per CDNA5 ISA section 8.3/8.4.
//   data_size_code: 0=1B 1=2B 2=4B 3=8B
//   pad_field: pre-encoded pad_enable/pad_interval/pad_amount bits (dw0)
// Issued once per wave (EXEC ignored); caller gates to a single wave.
// Toolchain here uses the 6-arg builtin: (g0, g1, g2, g3, g4, cpol).
// ---------------------------------------------------------------------------
static __device__ __forceinline__ void tdm_load_2d(
    unsigned lds_addr, const void* gptr, unsigned data_size_code,
    unsigned tile_d0, unsigned tile_d1,
    unsigned tensor_d0, unsigned tensor_d1,
    unsigned long long d0_stride, unsigned pad_field) {
  unsigned long long ga = (unsigned long long)(uintptr_t)gptr;
  u32x4 g0;
  g0.x = 1u;                                             // count=1, user mode
  g0.y = lds_addr;                                       // LDS byte address
  g0.z = (unsigned)(ga & 0xffffffffu);                   // global_addr[31:0]
  g0.w = (unsigned)((ga >> 32) & 0x01ffffffu) | (2u << 30);  // addr[56:32] | type=2
  i32x8 g1;
  g1[0] = (int)((data_size_code << 16) | pad_field);     // mask=0 | size | pad
  g1[1] = (int)((tensor_d0 & 0xffffu) << 16);            // barrier_addr=0 | dim0 lo
  g1[2] = (int)(((tensor_d0 >> 16) & 0xffffu) | ((tensor_d1 & 0xffffu) << 16));
  g1[3] = (int)(((tensor_d1 >> 16) & 0xffffu) | ((tile_d0 & 0xffffu) << 16));
  g1[4] = (int)(tile_d1 & 0xffffu);                      // tile_dim1 | tile_dim2=0
  g1[5] = (int)(unsigned)(d0_stride & 0xffffffffu);      // dim0_stride lo
  g1[6] = (int)((d0_stride >> 32) & 0xffffu);            // dim0_stride hi | dim1_stride=0
  g1[7] = 0;
  __builtin_amdgcn_tensor_load_to_lds(g0, g1, (i32x4)0, (i32x4)0, (i32x8)0, 0);
}
static __device__ __forceinline__ unsigned lds_addr_of(const void* p) {
  return (unsigned)(uintptr_t)p;   // generic LDS pointer: low 32 bits = LDS offset
}

// ---------------------------------------------------------------------------
// Pass 0a: convert weights fp32 -> bf16, fold biases.
// ---------------------------------------------------------------------------
__global__ __launch_bounds__(256) void prep_kernel(
    const float* __restrict__ Wih0, const float* __restrict__ Whh0,
    const float* __restrict__ Wih1, const float* __restrict__ Whh1,
    const float* __restrict__ bih0, const float* __restrict__ bhh0,
    const float* __restrict__ bih1, const float* __restrict__ bhh1,
    unsigned short* __restrict__ Wih0b, unsigned short* __restrict__ Whh0b,
    unsigned short* __restrict__ Wih1b, unsigned short* __restrict__ Whh1b,
    float* __restrict__ bias0, float* __restrict__ bias1) {
  int idx = blockIdx.x * 256 + threadIdx.x;
  if (idx < G4 * DD) Wih0b[idx] = f2bf(Wih0[idx]);
  if (idx < G4 * HH) {
    Whh0b[idx] = f2bf(Whh0[idx]);
    Wih1b[idx] = f2bf(Wih1[idx]);
    Whh1b[idx] = f2bf(Whh1[idx]);
  }
  if (idx < G4) {
    bias0[idx] = bih0[idx] + bhh0[idx];
    bias1[idx] = bih1[idx] + bhh1[idx];
  }
}

// ---------------------------------------------------------------------------
// Pass 0b: convert x fp32 -> bf16 (feeds the TDM-staged GEMM).
// ---------------------------------------------------------------------------
__global__ __launch_bounds__(256) void xconv_kernel(
    const float* __restrict__ x, unsigned short* __restrict__ xb) {
  size_t i = ((size_t)blockIdx.x * 256 + threadIdx.x) * 8;
  const float4* p = reinterpret_cast<const float4*>(x + i);
  float4 f0 = p[0], f1 = p[1];
  u32x4 u;
  u.x = pk2(f0.x, f0.y); u.y = pk2(f0.z, f0.w);
  u.z = pk2(f1.x, f1.y); u.w = pk2(f1.z, f1.w);
  *reinterpret_cast<u32x4*>(xb + i) = u;
}

// ---------------------------------------------------------------------------
// Pass 1: xp0[bt, 4H] = x[bt, :] @ Wih0^T + bias0     (M=16384, N=512, K=2048)
// Block: 512 threads = 16 waves, tile 64(M) x 512(N).
// A tiles (64 x 64 bf16) DMA'd into double-buffered LDS by the Tensor Data
// Mover, with hardware padding: interval = 32 DWORDs (one 128B row),
// amount = 4 DWORDs (16B) -> LDS row stride 72 halfs (bank-conflict pad).
// Inner loop is pure ds_load + global_load(B) + wmma.
// ---------------------------------------------------------------------------
#define PAD_A ((1u << 20) | (4u << 22) | (3u << 25))  // enable | ivl 32DW | pad 4DW

__global__ __launch_bounds__(512) void gemm_xp0_kernel(
    const unsigned short* __restrict__ xb, const unsigned short* __restrict__ Wb,
    const float* __restrict__ bias0, float* __restrict__ xp0) {
  __shared__ unsigned short As[2][64 * 72];  // double-buffered A tile

  const int tid  = threadIdx.x;
  const int wave = tid >> 5, lane = tid & 31;
  const int wm = wave >> 3;          // 0..1 : M sub-tile (32 rows)
  const int wn = wave & 7;           // 0..7 : N sub-tile (64 cols)
  const int lrow = lane & 15, lhi = lane >> 4;
  const int Mbase = blockIdx.x * 64;

  v8f acc[2][4];
#pragma unroll
  for (int ti = 0; ti < 2; ++ti)
#pragma unroll
    for (int j = 0; j < 4; ++j) acc[ti][j] = (v8f)0.0f;

  // preload first A tile via TDM
  if (wave == 0)
    tdm_load_2d(lds_addr_of(&As[0][0]), xb + (size_t)Mbase * DD,
                /*size*/1, /*tile*/64, 64, /*tensor*/DD, BB * TT,
                /*stride*/DD, PAD_A);
  __builtin_amdgcn_s_wait_tensorcnt(0);
  __syncthreads();

  for (int k0 = 0; k0 < DD; k0 += 64) {
    const int buf = (k0 >> 6) & 1;
    if (wave == 0 && (k0 + 64) < DD)   // prefetch next tile into other buffer
      tdm_load_2d(lds_addr_of(&As[buf ^ 1][0]), xb + (size_t)Mbase * DD + k0 + 64,
                  1, 64, 64, DD, BB * TT, DD, PAD_A);
#pragma unroll
    for (int kk = 0; kk < 64; kk += 32) {
      v16bf a0 = frag_ld(&As[buf][(wm * 32 +      lrow) * 72 + kk + lhi * 8]);
      v16bf a1 = frag_ld(&As[buf][(wm * 32 + 16 + lrow) * 72 + kk + lhi * 8]);
#pragma unroll
      for (int j = 0; j < 4; ++j) {
        const unsigned short* bp =
            Wb + (size_t)(wn * 64 + j * 16 + lrow) * DD + k0 + kk + lhi * 8;
        v16bf b = frag_ld(bp);
        acc[0][j] = wmma_bf16(a0, b, acc[0][j]);
        acc[1][j] = wmma_bf16(a1, b, acc[1][j]);
      }
    }
    __syncthreads();                        // all reads of As[buf] complete
    __builtin_amdgcn_s_wait_tensorcnt(0);   // prefetch landed (wave 0)
    __syncthreads();                        // new tile visible to all waves
  }

#pragma unroll
  for (int ti = 0; ti < 2; ++ti)
#pragma unroll
    for (int j = 0; j < 4; ++j) {
      int n = wn * 64 + j * 16 + lrow;
      float bv = bias0[n];
#pragma unroll
      for (int r = 0; r < 8; ++r) {
        int m = Mbase + wm * 32 + ti * 16 + r + lhi * 8;
        xp0[(size_t)m * G4 + n] = acc[ti][j][r] + bv;
      }
    }
}

// ---------------------------------------------------------------------------
// Pass 2: fused 2-layer LSTM recurrence + FC head.
// Grid: 4 blocks x 16 batch rows, 256 threads = 8 waves per block.
// Wave w owns hidden j-tile [16w,16w+16) and computes all 4 gate tiles, so
// the cell update is pure register math; c lives in VGPRs for all 256 steps.
// The [16 x 512] fp32 xp0 slice for step t+1 is TDM-prefetched into
// double-buffered LDS while step t computes. 48 wmma per wave per step.
// ---------------------------------------------------------------------------
__global__ __launch_bounds__(256) void lstm_rec_kernel(
    const float* __restrict__ xp0,
    const unsigned short* __restrict__ Whh0b,
    const unsigned short* __restrict__ Wih1b,
    const unsigned short* __restrict__ Whh1b,
    const float* __restrict__ bias1,
    const float* __restrict__ Wfc, const float* __restrict__ bfc,
    float* __restrict__ out) {
  __shared__ unsigned short h0s[16 * 136];  // 16 rows x (128+8 pad) halfs
  __shared__ unsigned short h1s[16 * 136];
  __shared__ float xps[2][16 * 512];        // double-buffered gate-bias slice

  const int tid  = threadIdx.x;
  const int wave = tid >> 5, lane = tid & 31;
  const int lrow = lane & 15, lhi = lane >> 4;
  const int j0    = wave * 16;        // hidden-column tile
  const int bbase = blockIdx.x * 16;  // batch rows of this block

  for (int i = tid; i < 16 * 136; i += 256) { h0s[i] = 0; h1s[i] = 0; }
  float c0r[8], c1r[8];
#pragma unroll
  for (int r = 0; r < 8; ++r) { c0r[r] = 0.0f; c1r[r] = 0.0f; }

  float bv1[4];
#pragma unroll
  for (int q = 0; q < 4; ++q) bv1[q] = bias1[q * HH + j0 + lrow];

  // prefetch xp0 slice for t=0: rows = batch (stride T*4H floats), cols = 4H
  if (wave == 0)
    tdm_load_2d(lds_addr_of(&xps[0][0]), xp0 + (size_t)bbase * TT * G4,
                /*size*/2, /*tile*/G4, 16, /*tensor*/G4, BB * TT,
                /*stride*/(unsigned long long)TT * G4, 0);
  __builtin_amdgcn_s_wait_tensorcnt(0);
  __syncthreads();

  for (int t = 0; t < TT; ++t) {
    const int cur = t & 1;
    if (wave == 0 && (t + 1) < TT)
      tdm_load_2d(lds_addr_of(&xps[cur ^ 1][0]),
                  xp0 + ((size_t)bbase * TT + (t + 1)) * G4,
                  2, G4, 16, G4, BB * TT, (unsigned long long)TT * G4, 0);

    // ---- layer 0: gates = xp0_slice + h0 @ Whh0^T --------------------------
    v16bf aH0[4];
#pragma unroll
    for (int ks = 0; ks < 4; ++ks)
      aH0[ks] = frag_ld(&h0s[lrow * 136 + ks * 32 + lhi * 8]);

    v8f acc[4];
#pragma unroll
    for (int q = 0; q < 4; ++q) {
      v8f a;
#pragma unroll
      for (int r = 0; r < 8; ++r)
        a[r] = xps[cur][(r + lhi * 8) * G4 + q * HH + j0 + lrow];
#pragma unroll
      for (int ks = 0; ks < 4; ++ks) {
        v16bf b = frag_ld(&Whh0b[(size_t)(q * HH + j0 + lrow) * HH + ks * 32 + lhi * 8]);
        a = wmma_bf16(aH0[ks], b, a);
      }
      acc[q] = a;
    }

    float h0new[8];
#pragma unroll
    for (int r = 0; r < 8; ++r) {
      float i_ = sigm(acc[0][r]);
      float f_ = sigm(acc[1][r]);
      float g_ = tanh_f(acc[2][r]);
      float o_ = sigm(acc[3][r]);
      float c  = f_ * c0r[r] + i_ * g_;
      c0r[r]   = c;
      h0new[r] = o_ * tanh_f(c);
    }
    __syncthreads();  // all reads of h0s and xps[cur] done

    // write new h0 (bf16); preload old-h1 fragments before overwrite
#pragma unroll
    for (int r = 0; r < 8; ++r)
      h0s[(r + lhi * 8) * 136 + j0 + lrow] = f2bf(h0new[r]);
    v16bf aH1[4];
#pragma unroll
    for (int ks = 0; ks < 4; ++ks)
      aH1[ks] = frag_ld(&h1s[lrow * 136 + ks * 32 + lhi * 8]);
    __syncthreads();  // new h0 visible; everyone's h1 reads done

    // ---- layer 1: gates = bias1 + h0_new @ Wih1^T + h1 @ Whh1^T ------------
    v16bf aH0n[4];
#pragma unroll
    for (int ks = 0; ks < 4; ++ks)
      aH0n[ks] = frag_ld(&h0s[lrow * 136 + ks * 32 + lhi * 8]);

#pragma unroll
    for (int q = 0; q < 4; ++q) {
      v8f a;
#pragma unroll
      for (int r = 0; r < 8; ++r) a[r] = bv1[q];
#pragma unroll
      for (int ks = 0; ks < 4; ++ks) {
        v16bf b = frag_ld(&Wih1b[(size_t)(q * HH + j0 + lrow) * HH + ks * 32 + lhi * 8]);
        a = wmma_bf16(aH0n[ks], b, a);
      }
#pragma unroll
      for (int ks = 0; ks < 4; ++ks) {
        v16bf b = frag_ld(&Whh1b[(size_t)(q * HH + j0 + lrow) * HH + ks * 32 + lhi * 8]);
        a = wmma_bf16(aH1[ks], b, a);
      }
      acc[q] = a;
    }

    float h1new[8];
#pragma unroll
    for (int r = 0; r < 8; ++r) {
      float i_ = sigm(acc[0][r]);
      float f_ = sigm(acc[1][r]);
      float g_ = tanh_f(acc[2][r]);
      float o_ = sigm(acc[3][r]);
      float c  = f_ * c1r[r] + i_ * g_;
      c1r[r]   = c;
      h1new[r] = o_ * tanh_f(c);
    }
    // safe: h1 readers finished before the previous barrier
#pragma unroll
    for (int r = 0; r < 8; ++r)
      h1s[(r + lhi * 8) * 136 + j0 + lrow] = f2bf(h1new[r]);

    __builtin_amdgcn_s_wait_tensorcnt(0);   // xp0 prefetch landed (wave 0)
    __syncthreads();                        // next slice visible to all
  }

  // ---- FC head: out[b] = h1[b,:] . Wfc + bfc ------------------------------
  if (tid < 16) {
    float s = 0.0f;
    for (int j = 0; j < HH; ++j) s += bf2f(h1s[tid * 136 + j]) * Wfc[j];
    out[bbase + tid] = s + bfc[0];
  }
}

// ---------------------------------------------------------------------------
// Launch
// ---------------------------------------------------------------------------
extern "C" void kernel_launch(void* const* d_in, const int* in_sizes, int n_in,
                              void* d_out, int out_size, void* d_ws, size_t ws_size,
                              hipStream_t stream) {
  const float* x    = (const float*)d_in[0];
  const float* Wih0 = (const float*)d_in[1];
  const float* Whh0 = (const float*)d_in[2];
  const float* bih0 = (const float*)d_in[3];
  const float* bhh0 = (const float*)d_in[4];
  const float* Wih1 = (const float*)d_in[5];
  const float* Whh1 = (const float*)d_in[6];
  const float* bih1 = (const float*)d_in[7];
  const float* bhh1 = (const float*)d_in[8];
  const float* Wfc  = (const float*)d_in[9];
  const float* bfc  = (const float*)d_in[10];
  float* out = (float*)d_out;

  // Workspace layout (16B aligned)
  char* ws = (char*)d_ws;
  float*          xp0   = (float*)ws;                          // 33,554,432 B
  unsigned short* xb    = (unsigned short*)(ws + 33554432);    // 67,108,864 B
  unsigned short* Wih0b = (unsigned short*)(ws + 100663296);   //  2,097,152 B
  unsigned short* Whh0b = (unsigned short*)(ws + 102760448);   //    131,072 B
  unsigned short* Wih1b = (unsigned short*)(ws + 102891520);   //    131,072 B
  unsigned short* Whh1b = (unsigned short*)(ws + 103022592);   //    131,072 B
  float*          bias0 = (float*)(ws + 103153664);            //      2,048 B
  float*          bias1 = (float*)(ws + 103155712);            //      2,048 B

  prep_kernel<<<(G4 * DD) / 256, 256, 0, stream>>>(
      Wih0, Whh0, Wih1, Whh1, bih0, bhh0, bih1, bhh1,
      Wih0b, Whh0b, Wih1b, Whh1b, bias0, bias1);

  xconv_kernel<<<(BB * TT * DD) / (256 * 8), 256, 0, stream>>>(x, xb);

  gemm_xp0_kernel<<<(BB * TT) / 64, 512, 0, stream>>>(xb, Wih0b, bias0, xp0);

  lstm_rec_kernel<<<BB / 16, 256, 0, stream>>>(
      xp0, Whh0b, Wih1b, Whh1b, bias1, Wfc, bfc, out);
}
